// AdaptiveMask_90503550861966
// MI455X (gfx1250) — matrix operand order repaired and checked by reading, not compiled, not run on hardware
//
#include <hip/hip_runtime.h>
#include <hip/hip_bf16.h>
#include <math.h>

// ---------------------------------------------------------------------------
// Bidirectional LSTM + linear/topk + LayerNorm + mask for MI455X (gfx1250).
// - bf16 WMMA (v_wmma_f32_16x16x32_bf16) for all GEMMs
// - recurrent weights register-resident for the whole scan
// - G[t] gate stream staged into LDS by the Tensor Data Mover (TDM),
//   double-buffered and synchronized with s_wait_tensorcnt.
// ---------------------------------------------------------------------------

typedef __bf16 bf16;
typedef __attribute__((ext_vector_type(16))) __bf16 v16bf;
typedef __attribute__((ext_vector_type(8)))  float  v8f;
typedef __attribute__((ext_vector_type(4)))  unsigned v4u;
typedef __attribute__((ext_vector_type(8)))  int      v8i;
typedef __attribute__((ext_vector_type(4)))  int      v4i;

#define B_   64
#define T_   1024
#define D_   512
#define H_   256
#define NG   1024   // 4*H (gate width per direction)
#define KTOP 20

__device__ __forceinline__ float sigm(float x) { return 1.0f / (1.0f + __expf(-x)); }

// ---------------------------------------------------------------------------
// TDM: DMA one 64x1024 bf16 tile (G[t], 128 KB) from global into LDS.
// D# per CDNA5 ISA:
//   group0: count=1 | lds_addr | global_addr[56:0] | type=2
//   group1: data_size=1(2B), tensor_dim0=1024, tensor_dim1=64,
//           tile_dim0=1024, tile_dim1=64, tensor_dim0_stride=1024
// Tracked with TENSORcnt (wave-level; EXEC ignored).
// ---------------------------------------------------------------------------
__device__ __forceinline__ void tdm_load_G(const bf16* gsrc, unsigned lds_byte_off) {
  unsigned long long ga = (unsigned long long)(size_t)gsrc;
  v4u g0 = { 1u,                                   // count=1, is_restore=0, gather=0
             lds_byte_off,                         // lds_addr (bytes)
             (unsigned)(ga & 0xFFFFFFFFu),         // global_addr[31:0]
             (unsigned)((ga >> 32) & 0x01FFFFFFu) | (2u << 30) };  // addr[56:32] | type=2
  v8i g1 = { (int)0x00010000u,        // data_size=1 (2 bytes); mask/flags/pad = 0
             (int)(1024u << 16),      // tensor_dim0[15:0] in [31:16]
             (int)(64u << 16),        // tensor_dim0[31:16]=0 | tensor_dim1[15:0]
             (int)(1024u << 16),      // tensor_dim1[31:16]=0 | tile_dim0=1024
             64,                      // tile_dim1=64 | tile_dim2=0
             1024,                    // tensor_dim0_stride[31:0]
             0,                       // stride0[47:32]=0 | stride1[15:0]=0
             0 };                     // stride1[47:16]=0
  v4i g2 = { 0, 0, 0, 0 };
  v4i g3 = { 0, 0, 0, 0 };
#if defined(__clang_major__) && (__clang_major__ >= 23)
  v8i g4 = { 0, 0, 0, 0, 0, 0, 0, 0 };
  __builtin_amdgcn_tensor_load_to_lds(g0, g1, g2, g3, g4, 0);
#else
  __builtin_amdgcn_tensor_load_to_lds(g0, g1, g2, g3, 0);
#endif
}

// ---------------------------------------------------------------------------
// Pre-swizzle x[B,T,D] (f32) into WMMA A-fragments (bf16), row r = t*64+b.
// ---------------------------------------------------------------------------
__global__ void pack_x_frags(const float* __restrict__ x, bf16* __restrict__ afrag) {
  size_t e = (size_t)blockIdx.x * blockDim.x + threadIdx.x;   // e < B*T*D
  int d = (int)(e % D_);
  size_t r = e / D_;
  int b = (int)(r % B_);
  int t = (int)(r / B_);
  float val = x[((size_t)b * T_ + t) * D_ + d];
  int mt = (int)(r >> 4), m = (int)(r & 15);
  int kf = d >> 5, kl = d & 31;
  int lane, v, s = kl & 1;
  if (kl < 8)       { lane = m;      v = kl >> 1;              }
  else if (kl < 16) { lane = m + 16; v = (kl - 8) >> 1;        }
  else if (kl < 24) { lane = m;      v = 4 + ((kl - 16) >> 1); }
  else              { lane = m + 16; v = 4 + ((kl - 24) >> 1); }
  afrag[(((size_t)mt * 16) + kf) * 512 + lane * 16 + v * 2 + s] = (bf16)val;
}

// ---------------------------------------------------------------------------
// Pre-swizzle weight W[N][K] (f32) into WMMA B-fragments (bf16), B[k][n]=W[n][k].
// ---------------------------------------------------------------------------
__global__ void pack_w_frags(const float* __restrict__ w, bf16* __restrict__ bfrag,
                             int N, int K) {
  size_t e = (size_t)blockIdx.x * blockDim.x + threadIdx.x;   // e < N*K
  int k = (int)(e % K);
  int n = (int)(e / K);
  int nt = n >> 4, nl = n & 15;
  int kf = k >> 5, kl = k & 31;
  int lane = nl + ((kl & 8) ? 16 : 0);
  int v = ((kl & 7) >> 1) + ((kl & 16) ? 4 : 0), s = kl & 1;
  int KF = K >> 5;
  bfrag[(((size_t)nt * KF) + kf) * 512 + lane * 16 + v * 2 + s] = (bf16)w[e];
}

// ---------------------------------------------------------------------------
// Kernel 1: G = X @ [W_ih_f | W_ih_b]^T + bias    (M=65536, N=2048, K=512)
// ---------------------------------------------------------------------------
__global__ void __launch_bounds__(256) gemm_input(const bf16* __restrict__ afrag,
                                                  const bf16* __restrict__ bfrag,
                                                  const float* __restrict__ b_f,
                                                  const float* __restrict__ b_b,
                                                  bf16* __restrict__ G) {
  const int lane = threadIdx.x & 31;
  const int w    = blockIdx.x * 8 + (threadIdx.x >> 5);
  const int mb   = w >> 5;     // 0..1023
  const int nb   = w & 31;     // 0..31

  v8f acc[4][4] = {};
  for (int kf = 0; kf < 16; ++kf) {
    v16bf a[4], bm[4];
    #pragma unroll
    for (int i = 0; i < 4; ++i)
      a[i] = *(const v16bf*)(afrag + (((size_t)(mb * 4 + i) * 16) + kf) * 512 + lane * 16);
    #pragma unroll
    for (int j = 0; j < 4; ++j)
      bm[j] = *(const v16bf*)(bfrag + (((size_t)(nb * 4 + j) * 16) + kf) * 512 + lane * 16);
    #pragma unroll
    for (int i = 0; i < 4; ++i)
      #pragma unroll
      for (int j = 0; j < 4; ++j)
        acc[i][j] = __builtin_amdgcn_wmma_f32_16x16x32_bf16(
            false, a[i], false, bm[j], (short)0, acc[i][j], false, false);
  }

  const int half8 = (lane >= 16) ? 8 : 0;
  const int nl = lane & 15;
  for (int i = 0; i < 4; ++i) {
    for (int j = 0; j < 4; ++j) {
      int n = (nb * 4 + j) * 16 + nl;
      const float* bias = (n < NG) ? b_f : b_b;
      int nc = (n < NG) ? n : n - NG;
      bf16* Gd = G + ((n < NG) ? (size_t)0 : (size_t)T_ * B_ * NG);
      float* ap = (float*)&acc[i][j];
      #pragma unroll
      for (int v = 0; v < 8; ++v) {
        int r = (mb * 4 + i) * 16 + v + half8;   // global row = t*64 + b
        int t = r >> 6, bb = r & 63;
        Gd[((size_t)t * B_ + bb) * NG + nc] = (bf16)(ap[v] + bias[nc]);
      }
    }
  }
}

// ---------------------------------------------------------------------------
// Kernel 2: sequential LSTM scan, one workgroup per direction (16 waves).
// W_hh register-resident; h in LDS A-fragment layout; G[t] staged by TDM
// into double-buffered LDS, overlapped with the WMMA phase of step t-1.
// LDS: hfrag 32KB @0 | gates0 128KB @32768 | gates1 128KB @163840  (288 KB)
// ---------------------------------------------------------------------------
__global__ void lstm_scan(const bf16* __restrict__ whhfrag,
                          const bf16* __restrict__ G,
                          float* __restrict__ lstm_o) {
  extern __shared__ char smem[];
  bf16* hfrag = (bf16*)smem;                                   // 32 KB
  bf16* gbuf0 = (bf16*)(smem + 32768);
  bf16* gbuf1 = (bf16*)(smem + 32768 + 131072);

  const int tid  = threadIdx.x;
  const int lane = tid & 31;
  const int wv   = tid >> 5;              // 0..15
  const int dir  = blockIdx.x;            // 0 = fwd, 1 = bwd
  const bf16* Gd = G + (size_t)dir * T_ * B_ * NG;

  // Load this wave's recurrent-weight fragments once; resident for all steps.
  v16bf wf[4][8];
  #pragma unroll
  for (int n = 0; n < 4; ++n)
    #pragma unroll
    for (int k = 0; k < 8; ++k)
      wf[n][k] = *(const v16bf*)(whhfrag + (size_t)dir * 64 * 8 * 512 +
                                 ((size_t)((wv * 4 + n) * 8 + k)) * 512 + lane * 16);

  // h0 = 0, c0 = 0
  for (int q = tid; q < 2048; q += 512) ((uint4*)hfrag)[q] = make_uint4(0, 0, 0, 0);
  float creg[32];
  #pragma unroll
  for (int q = 0; q < 32; ++q) creg[q] = 0.0f;

  const int brow = tid >> 3;              // batch row 0..63
  const int j0   = (tid & 7) * 32;        // hidden slice start

  // Kick off the TDM for step 0 (wave 0 only; EXEC-independent DMA).
  if (wv == 0) {
    const int t0 = dir ? (T_ - 1) : 0;
    tdm_load_G(Gd + (size_t)t0 * B_ * NG, 32768u);
  }
  __syncthreads();

  for (int step = 0; step < T_; ++step) {
    const int t   = dir ? (T_ - 1 - step) : step;
    const int cur = step & 1;
    bf16* gates = cur ? gbuf1 : gbuf0;

    // Wait for this step's TDM transfer, make it visible to all waves.
    if (wv == 0) __builtin_amdgcn_s_wait_tensorcnt(0);
    __syncthreads();

    // Issue the TDM for step+1 into the other buffer (overlaps compute).
    if (wv == 0 && step + 1 < T_) {
      const int tn = dir ? (T_ - 2 - step) : (step + 1);
      tdm_load_G(Gd + (size_t)tn * B_ * NG, cur ? 32768u : (32768u + 131072u));
    }
    // Warm L2 for step+2 (G is 256 MB, larger than the 192 MB L2).
    if (step + 2 < T_) {
      const int t2 = dir ? (T_ - 3 - step) : (step + 2);
      __builtin_prefetch((const char*)(Gd + (size_t)t2 * B_ * NG) + tid * 256, 0, 1);
    }

    // Phase B: acc = h @ W_hh^T  (A-frags from LDS, B-frags from registers)
    v8f acc[4][4] = {};   // [mtile][ntile-local]
    for (int kf = 0; kf < 8; ++kf) {
      v16bf a[4];
      #pragma unroll
      for (int mt = 0; mt < 4; ++mt)
        a[mt] = *(const v16bf*)(hfrag + ((size_t)(mt * 8 + kf)) * 512 + lane * 16);
      #pragma unroll
      for (int mt = 0; mt < 4; ++mt)
        #pragma unroll
        for (int n = 0; n < 4; ++n)
          acc[mt][n] = __builtin_amdgcn_wmma_f32_16x16x32_bf16(
              false, a[mt], false, wf[n][kf], (short)0, acc[mt][n], false, false);
    }

    // Phase C: gates += acc  (each lane owns disjoint (row,col) elements)
    {
      const int half8 = (lane >= 16) ? 8 : 0;
      const int nl = lane & 15;
      #pragma unroll
      for (int mt = 0; mt < 4; ++mt)
        #pragma unroll
        for (int n = 0; n < 4; ++n) {
          int col = (wv * 4 + n) * 16 + nl;
          float* ap = (float*)&acc[mt][n];
          #pragma unroll
          for (int v = 0; v < 8; ++v) {
            int idx = (mt * 16 + v + half8) * NG + col;
            gates[idx] = (bf16)((float)gates[idx] + ap[v]);
          }
        }
    }
    __syncthreads();

    // Phase D: pointwise LSTM cell; write h to global + repack into hfrag.
    {
      float hv[32];
      const bf16* gr = gates + brow * NG;
      #pragma unroll
      for (int q = 0; q < 32; ++q) {
        int j = j0 + q;
        float ig = sigm((float)gr[j]);
        float fg = sigm((float)gr[256 + j]);
        float gg = tanhf((float)gr[512 + j]);
        float og = sigm((float)gr[768 + j]);
        float c = fg * creg[q] + ig * gg;
        creg[q] = c;
        hv[q] = og * tanhf(c);
      }
      float* outp = lstm_o + ((size_t)brow * T_ + t) * D_ + dir * H_ + j0;
      #pragma unroll
      for (int q = 0; q < 32; ++q) outp[q] = hv[q];

      // A-fragment repack: row m, K-slice j0..j0+31 -> lanes {m, m+16}
      const int mt = brow >> 4, m = brow & 15, kf = j0 >> 5;
      bf16* fb = hfrag + ((size_t)(mt * 8 + kf)) * 512;
      #pragma unroll
      for (int q = 0; q < 8; ++q) {
        fb[m * 16 + q]            = (bf16)hv[q];        // K = 0..7
        fb[m * 16 + 8 + q]        = (bf16)hv[16 + q];   // K = 16..23
        fb[(m + 16) * 16 + q]     = (bf16)hv[8 + q];    // K = 8..15
        fb[(m + 16) * 16 + 8 + q] = (bf16)hv[24 + q];   // K = 24..31
      }
    }
    __syncthreads();
  }
}

// ---------------------------------------------------------------------------
// Kernel 3a: per (b,t) row — logit = dot(row, lin_w)+lin_b; LayerNorm in place.
// ---------------------------------------------------------------------------
__global__ void __launch_bounds__(256) row_ln_logit(float* __restrict__ lstm,
                                                    const float* __restrict__ lin_w,
                                                    const float* __restrict__ lin_b,
                                                    const float* __restrict__ ln_g,
                                                    const float* __restrict__ ln_bv,
                                                    float* __restrict__ logits) {
  __shared__ float sd[256];
  const int tid = threadIdx.x;
  const size_t rid = blockIdx.x;                 // rid = b*T + t
  float* row = lstm + rid * (size_t)D_;
  const float v0 = row[tid], v1 = row[tid + 256];

  sd[tid] = v0 * lin_w[tid] + v1 * lin_w[tid + 256];
  __syncthreads();
  for (int s = 128; s > 0; s >>= 1) { if (tid < s) sd[tid] += sd[tid + s]; __syncthreads(); }
  if (tid == 0) logits[rid] = sd[0] + lin_b[0];
  __syncthreads();

  sd[tid] = v0 + v1;
  __syncthreads();
  for (int s = 128; s > 0; s >>= 1) { if (tid < s) sd[tid] += sd[tid + s]; __syncthreads(); }
  const float mu = sd[0] * (1.0f / 512.0f);
  __syncthreads();

  const float d0 = v0 - mu, d1 = v1 - mu;
  sd[tid] = d0 * d0 + d1 * d1;
  __syncthreads();
  for (int s = 128; s > 0; s >>= 1) { if (tid < s) sd[tid] += sd[tid + s]; __syncthreads(); }
  const float inv = rsqrtf(sd[0] * (1.0f / 512.0f) + 1e-5f);

  row[tid]       = d0 * inv * ln_g[tid]       + ln_bv[tid];
  row[tid + 256] = d1 * inv * ln_g[tid + 256] + ln_bv[tid + 256];
}

// ---------------------------------------------------------------------------
// Kernel 3b: per batch — k=20 smallest logits (softmax is monotonic, so
// top_k(-softmax) == k smallest logits; ties -> lowest index), mask scatter.
// ---------------------------------------------------------------------------
__global__ void __launch_bounds__(256) topk_mask(const float* __restrict__ logits,
                                                 const unsigned char* __restrict__ mask_in,
                                                 int* __restrict__ idx_out,
                                                 float* __restrict__ mask_out) {
  __shared__ float vals[T_];
  __shared__ unsigned char hit[T_];
  __shared__ float rv[256];
  __shared__ int   ri[256];
  const int b = blockIdx.x, tid = threadIdx.x;

  for (int q = tid; q < T_; q += 256) {
    vals[q] = logits[(size_t)b * T_ + q];
    hit[q] = 0;
  }
  __syncthreads();

  for (int it = 0; it < KTOP; ++it) {
    float best = INFINITY; int bi = T_;
    for (int q = tid; q < T_; q += 256) {
      float v = vals[q];
      if (v < best || (v == best && q < bi)) { best = v; bi = q; }
    }
    rv[tid] = best; ri[tid] = bi;
    __syncthreads();
    for (int s = 128; s > 0; s >>= 1) {
      if (tid < s) {
        if (rv[tid + s] < rv[tid] ||
            (rv[tid + s] == rv[tid] && ri[tid + s] < ri[tid])) {
          rv[tid] = rv[tid + s]; ri[tid] = ri[tid + s];
        }
      }
      __syncthreads();
    }
    if (tid == 0) {
      int a = ri[0];
      idx_out[b * KTOP + it] = a;
      vals[a] = INFINITY;
      hit[a] = 1;
    }
    __syncthreads();
  }

  for (int q = tid; q < T_; q += 256)
    mask_out[(size_t)b * T_ + q] = (mask_in[(size_t)b * T_ + q] && !hit[q]) ? 1.0f : 0.0f;
}

// ---------------------------------------------------------------------------
// Host launcher.
// Workspace: Afrag 64MB | Bfrag_ih 2MB | Bfrag_hh 1MB | G 256MB | logits 256KB
// d_out: [indices B*20 (int bits)] [lstm_o_n B*T*D f32] [sgn_mask B*T 0/1 f32]
// ---------------------------------------------------------------------------
extern "C" void kernel_launch(void* const* d_in, const int* in_sizes, int n_in,
                              void* d_out, int out_size, void* d_ws, size_t ws_size,
                              hipStream_t stream) {
  (void)in_sizes; (void)n_in; (void)out_size; (void)ws_size;

  const float* x      = (const float*)d_in[0];
  const float* W_ih_f = (const float*)d_in[1];
  const float* W_hh_f = (const float*)d_in[2];
  const float* b_f    = (const float*)d_in[3];
  const float* W_ih_b = (const float*)d_in[4];
  const float* W_hh_b = (const float*)d_in[5];
  const float* b_b    = (const float*)d_in[6];
  const float* lin_w  = (const float*)d_in[7];
  const float* lin_b  = (const float*)d_in[8];
  const float* ln_g   = (const float*)d_in[9];
  const float* ln_bv  = (const float*)d_in[10];
  const unsigned char* mask = (const unsigned char*)d_in[11];

  char* ws = (char*)d_ws;
  bf16*  afrag    = (bf16*)ws;                                   // 67,108,864 B
  bf16*  bfrag_ih = (bf16*)(ws + (size_t)67108864);              //  2,097,152 B
  bf16*  bfrag_hh = (bf16*)(ws + (size_t)69206016);              //  1,048,576 B
  bf16*  G        = (bf16*)(ws + (size_t)70254592);              // 268,435,456 B
  float* logits   = (float*)(ws + (size_t)70254592 + 268435456); //    262,144 B

  float* out      = (float*)d_out;
  int*   idx_out  = (int*)d_out;                                 // B*KTOP ints
  float* lstm     = out + B_ * KTOP;                             // B*T*D f32
  float* mask_out = lstm + (size_t)B_ * T_ * D_;                 // B*T f32

  // Fragment pre-swizzle (one-time, parallel over all elements).
  pack_x_frags<<<(B_ * T_ * D_) / 256, 256, 0, stream>>>(x, afrag);
  pack_w_frags<<<(NG * D_) / 256, 256, 0, stream>>>(W_ih_f, bfrag_ih, NG, D_);
  pack_w_frags<<<(NG * D_) / 256, 256, 0, stream>>>(W_ih_b, bfrag_ih + (size_t)64 * 16 * 512, NG, D_);
  pack_w_frags<<<(NG * H_) / 256, 256, 0, stream>>>(W_hh_f, bfrag_hh, NG, H_);
  pack_w_frags<<<(NG * H_) / 256, 256, 0, stream>>>(W_hh_b, bfrag_hh + (size_t)64 * 8 * 512, NG, H_);

  // Chip-wide bf16 WMMA GEMM: pre-activation gates for both directions.
  gemm_input<<<4096, 256, 0, stream>>>(afrag, bfrag_ih, b_f, b_b, G);

  // Latency-critical recurrence: one WG per direction, 288 KB dynamic LDS
  // (hfrag 32KB + double-buffered TDM gate tiles 2x128KB).
  lstm_scan<<<2, 512, 294912, stream>>>(bfrag_hh, G, lstm);

  // Logits + LayerNorm (in place), then top-k + mask.
  row_ln_logit<<<B_ * T_, 256, 0, stream>>>(lstm, lin_w, lin_b, ln_g, ln_bv, logits);
  topk_mask<<<B_, 256, 0, stream>>>(logits, mask, idx_out, mask_out);
}
